// RecursiveAutoEncoderTorch_21698174779530
// MI455X (gfx1250) — matrix operand order repaired and checked by reading, not compiled, not run on hardware
//
#include <hip/hip_runtime.h>
#include <hip/hip_bf16.h>
#include <math.h>

// ---------------------------------------------------------------------------
// Problem constants (from the reference): L=8192 leaves, E=1024 embedding.
//   G[j] = W2 @ leaf[j+1] + b            (parallel GEMM, WMMA f16->f32)
//   u_j  = tanh( (W1 @ u_{j-1}) * inv||u_{j-1}|| + G[j] )   (sequential chain)
//   out[j] = u_j * inv||u_j||
// ---------------------------------------------------------------------------
#define L_LEAVES 8192
#define EMB      1024
#define NSTEP    (L_LEAVES - 1)      // 8191 internal nodes
#define GROWS    L_LEAVES            // G padded to 8192 rows -> branch-free stores
#define NWG      16                  // chain workgroups (cluster-sized)
#define ROWS_PER_WG (EMB / NWG)      // 64 output rows per WG

typedef __attribute__((ext_vector_type(16))) _Float16 v16h;
typedef __attribute__((ext_vector_type(8)))  float    v8f;

// ---------------------------------------------------------------------------
// Init: zero the global barrier state and seed the "previous u" buffer with
// the first child feats[k0-1] (leaf 0 for the given sKids), read generically.
// ---------------------------------------------------------------------------
__global__ void rae_init_kernel(const float* __restrict__ emb,
                                const long long* __restrict__ skids,
                                float* __restrict__ buf_init,
                                unsigned* __restrict__ bar) {
    int t = threadIdx.x;
    if (t < 2) bar[t] = 0u;  // [0]=arrive count, [1]=phase
    long long k0 = skids[(size_t)2 * L_LEAVES + 0];
    int idx = (int)(k0 - 1);
    if (idx < 0 || idx >= 2 * L_LEAVES - 1) idx = 0;
    buf_init[t] = emb[(size_t)idx * EMB + t];
}

// ---------------------------------------------------------------------------
// GEMM: G = Emb2 @ W2^T + b, where Emb2[j,:] = leaf[sKids[L+j,1]-1, :].
// One wave per 16x32 tile of G (one M-tile x two N-tiles, A reused for both);
// K-loop of 32 per v_wmma_f32_16x16x32_f16, unrolled x2.
// ---------------------------------------------------------------------------
__global__ __launch_bounds__(256) void rae_gemm_kernel(
    const float* __restrict__ emb,        // (8192, 1024) f32
    const float* __restrict__ encW,       // (1024, 2048) f32, W2 = [:,1024:]
    const float* __restrict__ encB,       // (1024,) f32
    const long long* __restrict__ skids,  // (2L-1, 2) i64
    float* __restrict__ G)                // (8192, 1024) f32 (row 8191 = pad)
{
    const int lane  = threadIdx.x & 31;
    const int wave  = threadIdx.x >> 5;          // 0..7
    const int l     = lane & 15;                 // row (A) / col (B,D)
    const int hi    = lane >> 4;                 // half-wave selector
    const int mtile = blockIdx.x * 8 + wave;     // 0..511  (rows of G)
    const int nt0   = blockIdx.y * 2;            // 0..62   (two N-tiles per wave)

    // A-matrix row for this lane: leaf index gathered through sKids.
    int j0 = mtile * 16 + l;
    int jj = j0 < NSTEP ? j0 : (NSTEP - 1);
    int leaf = (int)(skids[((size_t)(L_LEAVES + jj)) * 2 + 1] - 1);
    if (leaf < 0 || leaf >= L_LEAVES) leaf = 0;
    const float* arow = emb + (size_t)leaf * EMB;

    // B-matrix columns for this lane: rows of W2 (encW[:,1024:]).
    const int nc0 = nt0 * 16 + l;
    const int nc1 = nc0 + 16;
    const float* brow0 = encW + (size_t)nc0 * (2 * EMB) + EMB;
    const float* brow1 = encW + (size_t)nc1 * (2 * EMB) + EMB;

    v8f acc0 = {}, acc1 = {};
#pragma unroll 2
    for (int kk = 0; kk < EMB; kk += 32) {
        // A 16x32 f16 layout: lanes0-15 hold K=kk..kk+15,
        //                     lanes16-31 hold K=kk+16..kk+31.
        float af[16];
        *(float4*)&af[0]  = *(const float4*)(arow + kk + hi * 16);
        *(float4*)&af[4]  = *(const float4*)(arow + kk + hi * 16 + 4);
        *(float4*)&af[8]  = *(const float4*)(arow + kk + hi * 16 + 8);
        *(float4*)&af[12] = *(const float4*)(arow + kk + hi * 16 + 12);
        // B 32x16 f16 layout: lanes0-15 hold K=kk..kk+15, lanes16-31 K=kk+16..kk+31.
        float bf0[16], bf1[16];
        *(float4*)&bf0[0]  = *(const float4*)(brow0 + kk + hi * 16);
        *(float4*)&bf0[4]  = *(const float4*)(brow0 + kk + hi * 16 + 4);
        *(float4*)&bf0[8]  = *(const float4*)(brow0 + kk + hi * 16 + 8);
        *(float4*)&bf0[12] = *(const float4*)(brow0 + kk + hi * 16 + 12);
        *(float4*)&bf1[0]  = *(const float4*)(brow1 + kk + hi * 16);
        *(float4*)&bf1[4]  = *(const float4*)(brow1 + kk + hi * 16 + 4);
        *(float4*)&bf1[8]  = *(const float4*)(brow1 + kk + hi * 16 + 8);
        *(float4*)&bf1[12] = *(const float4*)(brow1 + kk + hi * 16 + 12);

        v16h av, bv0, bv1;
#pragma unroll
        for (int e = 0; e < 16; ++e) {
            av[e]  = (_Float16)af[e];
            bv0[e] = (_Float16)bf0[e];
            bv1[e] = (_Float16)bf1[e];
        }
        acc0 = __builtin_amdgcn_wmma_f32_16x16x32_f16(
            false, av, false, bv0, (short)0, acc0, false, false);
        acc1 = __builtin_amdgcn_wmma_f32_16x16x32_f16(
            false, av, false, bv1, (short)0, acc1, false, false);
    }

    // Branch-free epilogue: G is padded to 8192 rows, overflow row absorbed.
    const float bias0 = encB[nc0];
    const float bias1 = encB[nc1];
#pragma unroll
    for (int d = 0; d < 8; ++d) {
        size_t row = (size_t)(mtile * 16 + d + 8 * hi);
        G[row * EMB + nc0] = acc0[d] + bias0;
        G[row * EMB + nc1] = acc1[d] + bias1;
    }
}

// ---------------------------------------------------------------------------
// Grid-wide barrier for the chain kernel: agent-scope release fence, atomic
// sense-reversing barrier (portable), plus s_cluster_barrier (gfx1250 HW path
// when cluster-dispatched; architecturally NOP otherwise), acquire fence.
// ---------------------------------------------------------------------------
__device__ inline void grid_barrier(unsigned* bar, unsigned* localphase) {
    __builtin_amdgcn_fence(__ATOMIC_RELEASE, "agent");  // publish this wave's stores
    __syncthreads();
    if (threadIdx.x == 0) {
        unsigned ph = *localphase;
        unsigned prev = __hip_atomic_fetch_add(&bar[0], 1u, __ATOMIC_ACQ_REL,
                                               __HIP_MEMORY_SCOPE_AGENT);
        if (prev == (unsigned)(NWG - 1)) {
            __hip_atomic_store(&bar[0], 0u, __ATOMIC_RELAXED, __HIP_MEMORY_SCOPE_AGENT);
            __hip_atomic_store(&bar[1], ph + 1u, __ATOMIC_RELEASE, __HIP_MEMORY_SCOPE_AGENT);
        } else {
            while (__hip_atomic_load(&bar[1], __ATOMIC_RELAXED,
                                     __HIP_MEMORY_SCOPE_AGENT) <= ph) {
                __builtin_amdgcn_s_sleep(2);
            }
        }
        *localphase = ph + 1u;
    }
    __builtin_amdgcn_s_cluster_barrier();  // s_barrier_signal/-wait -3 (NOP if unclustered)
    __syncthreads();
    __builtin_amdgcn_fence(__ATOMIC_ACQUIRE, "agent");  // invalidate stale lines
}

// ---------------------------------------------------------------------------
// Sequential chain: 16 WGs, each owns 64 rows of W1 kept RESIDENT IN VGPRS
// (64 f32 per thread -> full 4MB W1 never re-read from memory). Per step:
// stage u_{j-1} (4KB) to LDS, each thread does a 64-MAC slice, a 16-lane
// butterfly reduces both the dot product AND ||u||^2 (the 16 column-groups
// of one row tile the full vector), then tanh + lazy normalization.
// ---------------------------------------------------------------------------
__global__ __launch_bounds__(1024, 1) void rae_chain_kernel(
    const float* __restrict__ encW,  // (1024, 2048): W1 = [:, :1024]
    const float* __restrict__ G,     // (8192, 1024), row 8191 = pad
    float* __restrict__ out,         // (8191, 1024)
    float* __restrict__ buf0,
    float* __restrict__ buf1,
    unsigned* __restrict__ bar)
{
    const int wg = blockIdx.x;           // 0..15
    const int t  = threadIdx.x;          // 0..1023
    const int row_local = t >> 4;        // 0..63
    const int cg = t & 15;               // column group (64 cols each)
    const int row_g = wg * ROWS_PER_WG + row_local;

    // W1 slice resident in registers.
    float w[64];
    {
        const float* wr = encW + (size_t)row_g * (2 * EMB) + cg * 64;
#pragma unroll
        for (int i = 0; i < 64; ++i) w[i] = wr[i];
    }

    __shared__ float u_ex[EMB];           // full previous u (staged)
    __shared__ float u_loc[ROWS_PER_WG];  // this WG's own u from last step

    unsigned localphase = 0;

    for (int j = 0; j < NSTEP; ++j) {
        const float* bufp = (j & 1) ? buf0 : buf1;   // previous u (j=0 -> init buf1)
        float*       bufc = (j & 1) ? buf1 : buf0;   // this step's u

        u_ex[t] = bufp[t];
        // Warm next step's G row into cache (row j+1 <= 8191 is in the pad).
        if (cg == 0)
            __builtin_prefetch(&G[(size_t)(j + 1) * EMB + row_g], 0, 0);
        __syncthreads();

        float sp = 0.f, sq = 0.f;
        const float* us = &u_ex[cg * 64];
#pragma unroll
        for (int i = 0; i < 64; ++i) {
            float u = us[i];
            sp = fmaf(w[i], u, sp);   // dot-product partial
            sq = fmaf(u, u, sq);      // ||u_{j-1}||^2 partial
        }
#pragma unroll
        for (int m = 1; m < 16; m <<= 1) {
            sp += __shfl_xor(sp, m, 32);
            sq += __shfl_xor(sq, m, 32);
        }

        if (cg == 0) {  // one leader per output row
            float invn = (j == 0) ? 1.0f : rsqrtf(sq);  // feats[0] is a raw leaf
            float y = fmaf(sp, invn, G[(size_t)j * EMB + row_g]);
            float u = tanhf(y);
            if (j > 0)  // emit step j-1 now that its norm is known
                out[(size_t)(j - 1) * EMB + row_g] = u_loc[row_local] * invn;
            u_loc[row_local] = u;
            bufc[row_g] = u;   // publish our 64 rows of u_j
        }
        grid_barrier(bar, &localphase);
    }

    // Epilogue: normalize and emit the final step.
    const float* bufl = ((NSTEP - 1) & 1) ? buf1 : buf0;
    u_ex[t] = bufl[t];
    __syncthreads();
    float sq = 0.f;
    const float* us = &u_ex[cg * 64];
#pragma unroll
    for (int i = 0; i < 64; ++i) { float u = us[i]; sq = fmaf(u, u, sq); }
#pragma unroll
    for (int m = 1; m < 16; m <<= 1) sq += __shfl_xor(sq, m, 32);
    if (cg == 0)
        out[(size_t)(NSTEP - 1) * EMB + row_g] = u_loc[row_local] * rsqrtf(sq);
}

// ---------------------------------------------------------------------------
extern "C" void kernel_launch(void* const* d_in, const int* in_sizes, int n_in,
                              void* d_out, int out_size, void* d_ws, size_t ws_size,
                              hipStream_t stream) {
    const float*     emb   = (const float*)d_in[0];      // (8192,1024)
    const float*     encW  = (const float*)d_in[1];      // (1024,2048)
    const float*     encB  = (const float*)d_in[2];      // (1024,)
    const long long* skids = (const long long*)d_in[3];  // (16383,2) i64
    float*           out   = (float*)d_out;              // (8191,1024)

    // Workspace layout: G (8192x1024 f32, 32MB) | buf0 (4KB) | buf1 (4KB) | barrier
    float*    G    = (float*)d_ws;
    float*    buf0 = G + (size_t)GROWS * EMB;
    float*    buf1 = buf0 + EMB;
    unsigned* bar  = (unsigned*)(buf1 + EMB);

    rae_init_kernel<<<dim3(1), dim3(EMB), 0, stream>>>(emb, skids, buf1, bar);

    dim3 ggrid(64, 32);  // 512 M-tiles x (32 blocks x 2 N-tiles/wave), 8 waves/block
    rae_gemm_kernel<<<ggrid, dim3(256), 0, stream>>>(emb, encW, encB, skids, G);

    rae_chain_kernel<<<dim3(NWG), dim3(1024), 0, stream>>>(encW, G, out, buf0, buf1, bar);
}
